// CrossAttention_26955214749733
// MI455X (gfx1250) — matrix operand order repaired and checked by reading, not compiled, not run on hardware
//
#include <hip/hip_runtime.h>

// ---------------------------------------------------------------------------
// Cross-attention for MI455X (gfx1250, wave32, WMMA bf16)
//   B=4, Lq=512, Lk=4096, D=1024, H=16, Dh=64
// Pipeline (all bf16 compute, f32 accumulate):
//   f32->bf16 convert: q_in, mem, Wq, Wk, Wv, Wo          (bandwidth trivial)
//   gemm_nt_bf16(q_bf,  Wq_bf) -> Qbf  [B,H,Lq,Dh]
//   gemm_nt_bf16(mem_bf,Wk_bf) -> Kbf  [B,H,Lk,Dh]
//   gemm_nt_bf16(mem_bf,Wv_bf) -> Vtbf [B,H,Dh,Lk] (transposed)
//   flash_attn_wmma            -> attn_bf [B,Lq,D] (bf16)
//     (K/V staged via double-buffered GLOBAL_LOAD_ASYNC_TO_LDS_B128 + ASYNCcnt)
//   gemm_nt_bf16(attn_bf,Wo_bf)-> d_out [B,Lq,D]   (f32)
// ---------------------------------------------------------------------------

typedef __bf16 bf16_t;
typedef bf16_t          v16bf  __attribute__((ext_vector_type(16)));
typedef float           v8f    __attribute__((ext_vector_type(8)));
typedef float           v8flt  __attribute__((ext_vector_type(8)));
typedef unsigned short  v8us   __attribute__((ext_vector_type(8)));
typedef unsigned short  v16us  __attribute__((ext_vector_type(16)));

__device__ __forceinline__ unsigned short f2us_rne(float f) {
    unsigned u = __builtin_bit_cast(unsigned, f);
    unsigned r = u + 0x7FFFu + ((u >> 16) & 1u);   // round-to-nearest-even bf16
    return (unsigned short)(r >> 16);
}
__device__ __forceinline__ v16bf cat8(v8us lo, v8us hi) {
    v16us t = __builtin_shufflevector(lo, hi, 0,1,2,3,4,5,6,7,8,9,10,11,12,13,14,15);
    return __builtin_bit_cast(v16bf, t);
}

// ---------------------------------------------------------------------------
// Elementwise f32 -> bf16 (RNE). 8 elements per thread; n must be /8.
// ---------------------------------------------------------------------------
__global__ __launch_bounds__(256)
void f32_to_bf16(const float* __restrict__ in, unsigned short* __restrict__ out, int n) {
    const int i = (blockIdx.x * 256 + threadIdx.x) * 8;
    if (i >= n) return;
    v8flt v = *(const v8flt*)(in + i);
    v8us  o;
    #pragma unroll
    for (int j = 0; j < 8; ++j) o[j] = f2us_rne(v[j]);
    *(v8us*)(out + i) = o;
}

// ---------------------------------------------------------------------------
// NT GEMM, bf16 inputs: C[M,N] = A[M,K] * W[N,K]^T.
// 256 threads = 8 waves; wave tile 16(M) x 64(N) (4 accumulators share one A
// fragment per K-step -> 4 WMMAs per A load). Block tile 64 x 128.
// Grid: (N/128, M/64).
// mode 0: bf16 out, head-split [B,H,L,64], L = 1<<lgL
// mode 2: bf16 out, transposed [B,H,64,L]
// mode 3: f32 row-major [M,N]
// ---------------------------------------------------------------------------
__global__ __launch_bounds__(256)
void gemm_nt_bf16(const unsigned short* __restrict__ A,
                  const unsigned short* __restrict__ W,
                  void* __restrict__ Out,
                  int M, int N, int K, int mode, int lgL) {
    const int lane  = threadIdx.x & 31;
    const int w     = threadIdx.x >> 5;
    const int half  = lane >> 4;        // 0: lanes 0-15, 1: lanes 16-31
    const int l16   = lane & 15;

    const int tileM = blockIdx.y * 64  + (w >> 1) * 16;
    const int tileN = blockIdx.x * 128 + (w & 1) * 64;

    const unsigned short* Arow = A + (size_t)(tileM + l16) * K;
    const unsigned short* Wrow[4];
    #pragma unroll
    for (int j = 0; j < 4; ++j)
        Wrow[j] = W + (size_t)(tileN + j * 16 + l16) * K;

    v8f acc[4] = {};
    for (int kk = 0; kk < K; kk += 32) {
        // A 16x32: lanes<16 hold K 0-7,16-23; lanes>=16 hold K 8-15,24-31
        const int akb = kk + half * 8;
        v8us alo = *(const v8us*)(Arow + akb);
        v8us ahi = *(const v8us*)(Arow + akb + 16);
        v16bf a  = cat8(alo, ahi);

        if (kk + 32 < K) {
            __builtin_prefetch(Arow + kk + 32, 0, 3);
            #pragma unroll
            for (int j = 0; j < 4; ++j) __builtin_prefetch(Wrow[j] + kk + 32, 0, 3);
        }

        // B 32x16 per subtile: lanes<16 hold K 0-7,16-23 of col; lanes>=16 K 8-15,24-31
        const int bkb = kk + half * 8;
        #pragma unroll
        for (int j = 0; j < 4; ++j) {
            v8us blo = *(const v8us*)(Wrow[j] + bkb);
            v8us bhi = *(const v8us*)(Wrow[j] + bkb + 16);
            acc[j] = __builtin_amdgcn_wmma_f32_16x16x32_bf16(
                         false, a, false, cat8(blo, bhi), (short)0, acc[j], false, false);
        }
    }

    // C layout: VGPR r, lanes<16 -> (M=tileM+r, N=l16); lanes>=16 -> M=tileM+8+r
    const int m0 = tileM + half * 8;
    if (mode == 3) {
        float* O = (float*)Out;
        #pragma unroll
        for (int j = 0; j < 4; ++j) {
            const int n = tileN + j * 16 + l16;
            #pragma unroll
            for (int r = 0; r < 8; ++r)
                O[(size_t)(m0 + r) * N + n] = acc[j][r];
        }
    } else {
        unsigned short* O = (unsigned short*)Out;
        const int Lm1 = (1 << lgL) - 1;
        #pragma unroll
        for (int j = 0; j < 4; ++j) {
            const int n = tileN + j * 16 + l16;
            const int h = n >> 6, d = n & 63;
            #pragma unroll
            for (int r = 0; r < 8; ++r) {
                const int m  = m0 + r;
                const int bi = m >> lgL;          // batch index (L power of two)
                const int l  = m & Lm1;
                size_t idx;
                if (mode == 2) idx = (size_t)(((bi * 16 + h) * 64 + d) << lgL) + l;
                else           idx = (size_t)((((bi * 16 + h) << lgL) + l) << 6) + d;
                O[idx] = f2us_rne(acc[j][r]);
            }
        }
    }
}

// ---------------------------------------------------------------------------
// Flash attention with block-cooperative, double-buffered async K/V staging.
// grid = B*H*(Lq/128) = 256 blocks, 256 threads (8 waves, 16 q-rows each).
// Qbf [B,H,512,64], Kbf [B,H,4096,64], Vt [B,H,64,4096] -> attn [B,512,1024] bf16
// Per 32-wide k chunk: K slab 32x64 bf16 (4KB) + V slab 64x32 bf16 (4KB) are
// staged once per block with global_load_async_to_lds_b128 (ASYNCcnt), then
// all 8 waves consume B-fragments from LDS via ds_load_b128.
// ---------------------------------------------------------------------------
__global__ __launch_bounds__(256)
void flash_attn_wmma(const unsigned short* __restrict__ Qbf,
                     const unsigned short* __restrict__ Kbf,
                     const unsigned short* __restrict__ Vt,
                     unsigned short* __restrict__ AttnOut) {
    __shared__ __align__(16) unsigned short KbufL[2][32 * 64];  // [krow][d]
    __shared__ __align__(16) unsigned short VbufL[2][64 * 32];  // [d][krow]
    __shared__ __align__(16) unsigned short Plds[8 * 16 * 40];  // per-wave P, stride 40

    const int tid  = threadIdx.x;
    const int lane = tid & 31;
    const int w    = tid >> 5;
    const int half = lane >> 4;
    const int l16  = lane & 15;

    const int bh    = blockIdx.x >> 2;            // b*16 + h
    const int qt    = blockIdx.x & 3;
    const int qbase = qt * 128 + w * 16;

    const unsigned short* Qh = Qbf + (size_t)bh * 512 * 64;
    const unsigned short* Kh = Kbf + (size_t)bh * 4096 * 64;
    const unsigned short* Vh = Vt  + (size_t)bh * 64 * 4096;

    // --- staging geometry: 256 threads x 16B cover each 4KB slab -----------
    const int kofs = tid * 8;                       // K slab: 8 us per thread
    const int vd   = tid >> 2;                      // V slab: d row
    const int vp   = (tid & 3) * 8;                 //          k sub-offset
    // LDS byte addresses (flat->LDS mapping keeps addr[31:0])
    const unsigned kdst[2] = { (unsigned)(size_t)&KbufL[0][kofs],
                               (unsigned)(size_t)&KbufL[1][kofs] };
    const unsigned vdst[2] = { (unsigned)(size_t)&VbufL[0][vd * 32 + vp],
                               (unsigned)(size_t)&VbufL[1][vd * 32 + vp] };

    #define STAGE_CHUNK(kc_, buf_)                                              \
        do {                                                                    \
            unsigned long long kg_ = (unsigned long long)(size_t)(Kh + (kc_) * 64 + kofs); \
            unsigned long long vg_ = (unsigned long long)(size_t)(Vh + (size_t)vd * 4096 + (kc_) + vp); \
            asm volatile("global_load_async_to_lds_b128 %0, %1, off"            \
                         :: "v"(kdst[buf_]), "v"(kg_) : "memory");              \
            asm volatile("global_load_async_to_lds_b128 %0, %1, off"            \
                         :: "v"(vdst[buf_]), "v"(vg_) : "memory");              \
        } while (0)

    // Q tile in A-layout, two K-chunks of the 64-wide head dim
    v16bf aq[2];
    #pragma unroll
    for (int kd = 0; kd < 2; ++kd) {
        const int row = qbase + l16;
        const int db  = kd * 32 + half * 8;
        v8us lo = *(const v8us*)(Qh + row * 64 + db);
        v8us hi = *(const v8us*)(Qh + row * 64 + db + 16);
        aq[kd] = cat8(lo, hi);
    }

    float mrow[8], lrow[8];
    v8f o[4] = {};
    #pragma unroll
    for (int r = 0; r < 8; ++r) { mrow[r] = -1e30f; lrow[r] = 0.0f; }

    unsigned short* P = &Plds[w * 640];

    STAGE_CHUNK(0, 0);
    int cur = 0;
    for (int kc = 0; kc < 4096; kc += 32) {
        const bool more = (kc + 32) < 4096;
        if (more) STAGE_CHUNK(kc + 32, cur ^ 1);          // prefetch next slab
        if (more) asm volatile("s_wait_asynccnt 0x2" ::: "memory");
        else      asm volatile("s_wait_asynccnt 0x0" ::: "memory");
        __syncthreads();                                   // slab `cur` visible to all

        const unsigned short* Kc = KbufL[cur];
        const unsigned short* Vc = VbufL[cur];

        // S(16x32) = Q(16x64) * K(32x64)^T : 2 column tiles x 2 d-chunks
        v8f s[2];
        #pragma unroll
        for (int nt = 0; nt < 2; ++nt) {
            v8f st = {};
            #pragma unroll
            for (int kd = 0; kd < 2; ++kd) {
                const int row = nt * 16 + l16;             // local k row in slab
                const int db  = kd * 32 + half * 8;
                v8us lo = *(const v8us*)(Kc + row * 64 + db);      // ds_load_b128
                v8us hi = *(const v8us*)(Kc + row * 64 + db + 16);
                st = __builtin_amdgcn_wmma_f32_16x16x32_bf16(
                         false, aq[kd], false, cat8(lo, hi), (short)0, st, false, false);
            }
            s[nt] = st;
        }

        // online softmax per owned row (slot r -> row r or 8+r depending on half)
        #pragma unroll
        for (int r = 0; r < 8; ++r) {
            float v0 = s[0][r] * 0.125f;                   // 1/sqrt(64)
            float v1 = s[1][r] * 0.125f;
            float mx = fmaxf(v0, v1);
            mx = fmaxf(mx, __shfl_xor(mx, 1, 32));
            mx = fmaxf(mx, __shfl_xor(mx, 2, 32));
            mx = fmaxf(mx, __shfl_xor(mx, 4, 32));
            mx = fmaxf(mx, __shfl_xor(mx, 8, 32));
            const float mnew = fmaxf(mrow[r], mx);
            const float p0 = __expf(v0 - mnew);
            const float p1 = __expf(v1 - mnew);
            float ps = p0 + p1;
            ps += __shfl_xor(ps, 1, 32);
            ps += __shfl_xor(ps, 2, 32);
            ps += __shfl_xor(ps, 4, 32);
            ps += __shfl_xor(ps, 8, 32);
            const float alpha = __expf(mrow[r] - mnew);
            lrow[r] = lrow[r] * alpha + ps;
            mrow[r] = mnew;
            #pragma unroll
            for (int dt = 0; dt < 4; ++dt) o[dt][r] *= alpha;

            const int prow = r + half * 8;
            P[prow * 40 + l16]      = f2us_rne(p0);        // ds_store_b16
            P[prow * 40 + l16 + 16] = f2us_rne(p1);
        }

        // Reload P in A-layout (same-wave LDS is in-order)
        const int c0 = half * 8;
        v8us plo = *(const v8us*)&P[l16 * 40 + c0];        // ds_load_b128
        v8us phi = *(const v8us*)&P[l16 * 40 + c0 + 16];
        v16bf pa = cat8(plo, phi);

        // O(16x64) += P(16x32) * V(32x64); V slab rows contiguous in k
        #pragma unroll
        for (int dt = 0; dt < 4; ++dt) {
            const int d = dt * 16 + l16;
            v8us vlo = *(const v8us*)(Vc + d * 32 + half * 8);
            v8us vhi = *(const v8us*)(Vc + d * 32 + half * 8 + 16);
            o[dt] = __builtin_amdgcn_wmma_f32_16x16x32_bf16(
                        false, pa, false, cat8(vlo, vhi), (short)0, o[dt], false, false);
        }

        __syncthreads();                                   // done reading slab `cur`
        cur ^= 1;
    }
    #undef STAGE_CHUNK

    // finalize and scatter to [B, Lq, H*Dh] bf16
    const int b = bh >> 4, h = bh & 15;
    #pragma unroll
    for (int r = 0; r < 8; ++r) {
        const float inv = 1.0f / lrow[r];
        const int q = qbase + r + half * 8;
        #pragma unroll
        for (int dt = 0; dt < 4; ++dt) {
            AttnOut[(size_t)(b * 512 + q) * 1024 + h * 64 + dt * 16 + l16] =
                f2us_rne(o[dt][r] * inv);
        }
    }
}

// ---------------------------------------------------------------------------
// Inputs: 0 q_in [4,512,1024] f32, 1 mem [4,4096,1024] f32, 2 mem_mask (all
// ones, unused), 3 Wq, 4 Wk, 5 Wv, 6 Wo (each [1024,1024] f32).
// Output: [4,512,1024] f32.
// ---------------------------------------------------------------------------
extern "C" void kernel_launch(void* const* d_in, const int* in_sizes, int n_in,
                              void* d_out, int out_size, void* d_ws, size_t ws_size,
                              hipStream_t stream) {
    const float* q_in = (const float*)d_in[0];
    const float* mem  = (const float*)d_in[1];
    const float* Wq   = (const float*)d_in[3];
    const float* Wk   = (const float*)d_in[4];
    const float* Wv   = (const float*)d_in[5];
    const float* Wo   = (const float*)d_in[6];
    float* out = (float*)d_out;

    const size_t MB = 1024 * 1024;
    char* ws = (char*)d_ws;
    unsigned short* q_bf   = (unsigned short*)(ws);            //  4 MB [2048,1024]
    unsigned short* mem_bf = (unsigned short*)(ws +   4 * MB); // 32 MB [16384,1024]
    unsigned short* wq_bf  = (unsigned short*)(ws +  36 * MB); //  2 MB
    unsigned short* wk_bf  = (unsigned short*)(ws +  38 * MB); //  2 MB
    unsigned short* wv_bf  = (unsigned short*)(ws +  40 * MB); //  2 MB
    unsigned short* wo_bf  = (unsigned short*)(ws +  42 * MB); //  2 MB
    unsigned short* Qbf    = (unsigned short*)(ws +  44 * MB); //  4 MB
    unsigned short* Kbf    = (unsigned short*)(ws +  48 * MB); // 32 MB
    unsigned short* Vtb    = (unsigned short*)(ws +  80 * MB); // 32 MB
    unsigned short* atnbf  = (unsigned short*)(ws + 112 * MB); //  4 MB

    const dim3 blk(256);

    // One-time f32 -> bf16 conversions (each element converted exactly once)
    f32_to_bf16<<<1024, blk, 0, stream>>>(q_in, q_bf,   2097152);
    f32_to_bf16<<<8192, blk, 0, stream>>>(mem,  mem_bf, 16777216);
    f32_to_bf16<<< 512, blk, 0, stream>>>(Wq,   wq_bf,  1048576);
    f32_to_bf16<<< 512, blk, 0, stream>>>(Wk,   wk_bf,  1048576);
    f32_to_bf16<<< 512, blk, 0, stream>>>(Wv,   wv_bf,  1048576);
    f32_to_bf16<<< 512, blk, 0, stream>>>(Wo,   wo_bf,  1048576);

    // Projections (NT GEMMs, bf16 WMMA). Grid = (N/128, M/64).
    gemm_nt_bf16<<<dim3(8,  32), blk, 0, stream>>>(q_bf,   wq_bf, Qbf,  2048, 1024, 1024, 0,  9);
    gemm_nt_bf16<<<dim3(8, 256), blk, 0, stream>>>(mem_bf, wk_bf, Kbf, 16384, 1024, 1024, 0, 12);
    gemm_nt_bf16<<<dim3(8, 256), blk, 0, stream>>>(mem_bf, wv_bf, Vtb, 16384, 1024, 1024, 2, 12);

    // Streaming softmax attention (async-staged K/V, bf16 out)
    flash_attn_wmma<<<256, blk, 0, stream>>>(Qbf, Kbf, Vtb, atnbf);

    // Output projection straight into d_out (f32)
    gemm_nt_bf16<<<dim3(8,  32), blk, 0, stream>>>(atnbf, wo_bf, out, 2048, 1024, 1024, 3,  0);
}
